// QwenAttention_57269093925328
// MI455X (gfx1250) — compile-verified, matrix operand order checked by reading, not run on hardware
//
#include <hip/hip_runtime.h>
#include <stdint.h>

typedef __bf16 bf16_t;
typedef __attribute__((ext_vector_type(16))) __bf16 v16bf;
typedef __attribute__((ext_vector_type(8)))  __bf16 v8bf;
typedef __attribute__((ext_vector_type(8)))  float  v8f;
typedef __attribute__((ext_vector_type(4)))  unsigned int v4u;
typedef __attribute__((ext_vector_type(8)))  int v8i;
typedef __attribute__((ext_vector_type(4)))  int v4i;

#define S_LEN 2048
#define NH    32
#define NKV   8
#define HD    128
#define HSZ   4096
#define NQKV  6144            // (NH + 2*NKV) * HD
#define VT_LD (S_LEN + 32)    // padded so last causal key block never reads OOB

// ---------------------------------------------------------------------------
// WMMA helper: D = A(16x32 bf16) * B(32x16 bf16) + C(16x16 f32)
// ---------------------------------------------------------------------------
static __device__ __forceinline__ v8f wmma_bf16(v16bf a, v16bf b, v8f c) {
  return __builtin_amdgcn_wmma_f32_16x16x32_bf16(false, a, false, b, (short)0, c,
                                                 false, false);
}

// Load a 16x32 bf16 fragment from a row-major matrix (ld = row stride in elems).
// ISA layout: lanes 0-15 hold row (row0+lane), K {0..7,16..23};
// lanes 16-31 hold row (row0+lane-16), K {8..15,24..31}.  Two b128 loads/lane.
static __device__ __forceinline__ v16bf load_frag(const bf16_t* base, int row0,
                                                  int k0, int ld) {
  int l   = threadIdx.x & 31;
  int hlf = (l >> 4) & 1;
  const bf16_t* p = base + (size_t)(row0 + (l & 15)) * ld + k0 + hlf * 8;
  v8bf lo = *(const v8bf*)(p);
  v8bf hi = *(const v8bf*)(p + 16);
  return __builtin_shufflevector(lo, hi, 0, 1, 2, 3, 4, 5, 6, 7,
                                 8, 9, 10, 11, 12, 13, 14, 15);
}

// ---------------------------------------------------------------------------
// Tensor Data Mover: 2D tile load (rows x cols of bf16) global -> LDS.
// D# built per CDNA5 ISA (group0 + group1, groups 2/3
// zero for <=2D tensors).  data_size=1 (2 bytes).  Issued once per wave,
// tracked with TENSORcnt.  6-arg builtin form (clang-23 toolchain).
// ---------------------------------------------------------------------------
static __device__ __forceinline__ void tdm_load_2d(unsigned int lds_off,
                                                   const bf16_t* gptr,
                                                   unsigned int tensor_d0,
                                                   unsigned int tensor_d1,
                                                   unsigned int tile_d0,
                                                   unsigned int tile_d1,
                                                   unsigned long long stride0) {
  unsigned long long ga = (unsigned long long)(uintptr_t)gptr;
  v4u g0;
  g0.x = 1u;                                        // count=1, user mode
  g0.y = lds_off;                                   // lds_addr [63:32]
  g0.z = (unsigned int)ga;                          // global_addr[31:0]
  g0.w = (unsigned int)((ga >> 32) & 0x01FFFFFFu)   // global_addr[56:32]
       | 0x80000000u;                               // type=2 ("image")
  v8i g1;
  g1[0] = (int)(1u << 16);                          // wg_mask=0, data_size=1(2B)
  g1[1] = (int)((tensor_d0 & 0xFFFFu) << 16);       // tensor_dim0[15:0]
  g1[2] = (int)(((tensor_d0 >> 16) & 0xFFFFu) | ((tensor_d1 & 0xFFFFu) << 16));
  g1[3] = (int)(((tensor_d1 >> 16) & 0xFFFFu) | ((tile_d0 & 0xFFFFu) << 16));
  g1[4] = (int)(tile_d1 & 0xFFFFu);                 // tile_dim1; tile_dim2=0
  g1[5] = (int)(unsigned int)(stride0 & 0xFFFFFFFFull);   // dim0_stride[31:0]
  g1[6] = (int)(unsigned int)((stride0 >> 32) & 0xFFFFull); // stride[47:32]
  g1[7] = 0;
  v4i z4 = {0, 0, 0, 0};
  v8i z8 = {0, 0, 0, 0, 0, 0, 0, 0};
  __builtin_amdgcn_tensor_load_to_lds(g0, g1, z4, z4, z8, 0);
}

static __device__ __forceinline__ unsigned int lds_offset_of(const void* p) {
  // generic pointer to __shared__: low 32 bits are the wave-relative LDS offset
  return (unsigned int)(uintptr_t)p;
}

// ---------------------------------------------------------------------------
// fp32 -> bf16 conversion
// ---------------------------------------------------------------------------
__global__ void cvt_f32_bf16(const float* __restrict__ src,
                             bf16_t* __restrict__ dst, int n) {
  int i = blockIdx.x * blockDim.x + threadIdx.x;
  if (i < n) dst[i] = (bf16_t)src[i];
}

// ---------------------------------------------------------------------------
// NT GEMM: C[m,n] = sum_k A[m,k] * B[n,k]   (A: MxK, B: NxK, both bf16)
// Block (256 thr, 8 waves) computes 128x128; wave computes 32x64.
// A/B 128x32 tiles staged in LDS by the Tensor Data Mover, double-buffered.
// ---------------------------------------------------------------------------
template <bool OUT_F32>
__global__ __launch_bounds__(256) void gemm_nt(const bf16_t* __restrict__ A,
                                               const bf16_t* __restrict__ B,
                                               float* __restrict__ Cf,
                                               bf16_t* __restrict__ Cb,
                                               int M, int N, int K) {
  __shared__ __align__(16) bf16_t ldsA[2][128 * 32];
  __shared__ __align__(16) bf16_t ldsB[2][128 * 32];

  int n_blk = blockIdx.x * 128;
  int m_blk = blockIdx.y * 128;
  int w     = threadIdx.x >> 5;
  int msub  = (w >> 1) * 32;   // 4 wave-rows
  int nsub  = (w & 1) * 64;    // 2 wave-cols
  int l     = threadIdx.x & 31;
  int hlf   = l >> 4;
  int col   = l & 15;
  bool issuer = (w == 0);

  v8f acc[2][4];
#pragma unroll
  for (int mf = 0; mf < 2; mf++)
#pragma unroll
    for (int nf = 0; nf < 4; nf++) acc[mf][nf] = (v8f){0, 0, 0, 0, 0, 0, 0, 0};

  if (issuer) {
    tdm_load_2d(lds_offset_of(&ldsA[0][0]), A + (size_t)m_blk * K, K, M, 32,
                128, (unsigned long long)K);
    tdm_load_2d(lds_offset_of(&ldsB[0][0]), B + (size_t)n_blk * K, K, N, 32,
                128, (unsigned long long)K);
  }

  int buf = 0;
  for (int k0 = 0; k0 < K; k0 += 32, buf ^= 1) {
    bool more = (k0 + 32) < K;
    if (issuer) {
      if (more) {
        tdm_load_2d(lds_offset_of(&ldsA[buf ^ 1][0]),
                    A + (size_t)m_blk * K + (k0 + 32), K, M, 32, 128,
                    (unsigned long long)K);
        tdm_load_2d(lds_offset_of(&ldsB[buf ^ 1][0]),
                    B + (size_t)n_blk * K + (k0 + 32), K, N, 32, 128,
                    (unsigned long long)K);
        __builtin_amdgcn_s_wait_tensorcnt(2);  // current pair landed (in-order)
      } else {
        __builtin_amdgcn_s_wait_tensorcnt(0);
      }
    }
    __syncthreads();

    v16bf a0 = load_frag(&ldsA[buf][0], msub, 0, 32);
    v16bf a1 = load_frag(&ldsA[buf][0], msub + 16, 0, 32);
#pragma unroll
    for (int nf = 0; nf < 4; nf++) {
      v16bf b = load_frag(&ldsB[buf][0], nsub + nf * 16, 0, 32);
      acc[0][nf] = wmma_bf16(a0, b, acc[0][nf]);
      acc[1][nf] = wmma_bf16(a1, b, acc[1][nf]);
    }
    __syncthreads();
  }

#pragma unroll
  for (int mf = 0; mf < 2; mf++) {
#pragma unroll
    for (int nf = 0; nf < 4; nf++) {
#pragma unroll
      for (int r = 0; r < 8; r++) {
        float v = acc[mf][nf][r];
        int row = m_blk + msub + mf * 16 + r + 8 * hlf;
        int c   = n_blk + nsub + nf * 16 + col;
        if (OUT_F32) Cf[(size_t)row * N + c] = v;
        else         Cb[(size_t)row * N + c] = (bf16_t)v;
      }
    }
  }
}

// ---------------------------------------------------------------------------
// Per-head RMSNorm + RoPE for Q and K.  One wave per (s, head) row of 128.
// qkv: [S, 6144] bf16 (Q cols 0..4095, K cols 4096..5119)
// ---------------------------------------------------------------------------
__global__ __launch_bounds__(256) void norm_rope(const bf16_t* __restrict__ qkv,
                                                 const float* __restrict__ qw,
                                                 const float* __restrict__ kw,
                                                 const int* __restrict__ pos,
                                                 bf16_t* __restrict__ Qn,
                                                 bf16_t* __restrict__ Kn) {
  int wid  = (blockIdx.x * blockDim.x + threadIdx.x) >> 5;
  int lane = threadIdx.x & 31;
  int total = S_LEN * (NH + NKV);
  if (wid >= total) return;
  int s = wid / (NH + NKV);
  int c = wid % (NH + NKV);

  const bf16_t* src = qkv + (size_t)s * NQKV + c * HD;
  float x[4], ss = 0.f;
#pragma unroll
  for (int j = 0; j < 4; j++) {
    x[j] = (float)src[lane + 32 * j];
    ss += x[j] * x[j];
  }
  ss += __shfl_xor(ss, 1, 32);
  ss += __shfl_xor(ss, 2, 32);
  ss += __shfl_xor(ss, 4, 32);
  ss += __shfl_xor(ss, 8, 32);
  ss += __shfl_xor(ss, 16, 32);
  float scl = rsqrtf(ss * (1.0f / 128.0f) + 1e-5f);

  const float* w = (c < NH) ? qw : kw;
  float xn[4];
#pragma unroll
  for (int j = 0; j < 4; j++) xn[j] = x[j] * scl * w[lane + 32 * j];

  float p = (float)pos[s];
  const float l2t_over64 = 18.931568569324174f / 64.0f;  // log2(500000)/64
  float out[4];
#pragma unroll
  for (int j = 0; j < 4; j++) {
    int d  = lane + 32 * j;
    int dh = d & 63;
    float inv = exp2f(-(float)dh * l2t_over64);
    float ang = p * inv;
    float sn = __sinf(ang);
    float cs = __cosf(ang);
    float other = xn[j ^ 2];  // element at d ^ 64 lives in same lane, j^2
    out[j] = (d < 64) ? (xn[j] * cs - other * sn)
                      : (xn[j] * cs + other * sn);
  }

  bf16_t* dst = (c < NH) ? (Qn + (size_t)s * HSZ + c * HD)
                         : (Kn + (size_t)s * (NKV * HD) + (c - NH) * HD);
#pragma unroll
  for (int j = 0; j < 4; j++) dst[lane + 32 * j] = (bf16_t)out[j];
}

// ---------------------------------------------------------------------------
// V transpose: Vt[g][d][s] (ld = VT_LD, zero-padded tail) from qkv [S,6144]
// ---------------------------------------------------------------------------
__global__ __launch_bounds__(256) void v_transpose(const bf16_t* __restrict__ qkv,
                                                   bf16_t* __restrict__ Vt) {
  int s = blockIdx.x * 256 + threadIdx.x;
  int d = blockIdx.y;
  int g = blockIdx.z;
  if (s >= VT_LD) return;
  bf16_t v = (bf16_t)0.0f;
  if (s < S_LEN)
    v = qkv[(size_t)s * NQKV + (NH + NKV) * HD + g * HD + d];
  Vt[((size_t)g * HD + d) * VT_LD + s] = v;
}

// ---------------------------------------------------------------------------
// Causal GQA flash attention.  One wave per (head, 16-query tile).
// Qn: [S,4096], Kn: [S,1024], Vt: [NKV][128][VT_LD], O: [S,4096] (all bf16)
// ---------------------------------------------------------------------------
__global__ __launch_bounds__(32) void attn_fwd(const bf16_t* __restrict__ Qn,
                                               const bf16_t* __restrict__ Kn,
                                               const bf16_t* __restrict__ Vt,
                                               bf16_t* __restrict__ O) {
  __shared__ __align__(16) bf16_t pbuf[16 * 32];

  int m0 = blockIdx.x * 16;
  int h  = blockIdx.y;
  int g  = h % NKV;  // GQA: q head h -> kv head h % KV
  int l   = threadIdx.x & 31;
  int hlf = l >> 4;
  int col = l & 15;

  const bf16_t* Qh = Qn + h * HD;
  const bf16_t* Kh = Kn + g * HD;
  const bf16_t* Vh = Vt + (size_t)g * HD * VT_LD;

  v8f acc[8];
#pragma unroll
  for (int f = 0; f < 8; f++) acc[f] = (v8f){0, 0, 0, 0, 0, 0, 0, 0};
  float mrun[8], lrun[8];
#pragma unroll
  for (int r = 0; r < 8; r++) { mrun[r] = -1e30f; lrun[r] = 0.0f; }

  const float scale = 0.08838834764831845f;  // 1/sqrt(128)

  for (int kb = 0; kb < m0 + 16; kb += 32) {
    bool do2 = kb < m0;
    v8f s0 = (v8f){0, 0, 0, 0, 0, 0, 0, 0};
    v8f s1 = (v8f){0, 0, 0, 0, 0, 0, 0, 0};
#pragma unroll
    for (int dk = 0; dk < HD; dk += 32) {
      v16bf a  = load_frag(Qh, m0, dk, HSZ);
      v16bf b0 = load_frag(Kh, kb, dk, NKV * HD);
      s0 = wmma_bf16(a, b0, s0);
      if (do2) {
        v16bf b1 = load_frag(Kh, kb + 16, dk, NKV * HD);
        s1 = wmma_bf16(a, b1, s1);
      }
    }

#pragma unroll
    for (int r = 0; r < 8; r++) {
      int q   = m0 + r + 8 * hlf;
      int k0i = kb + col;
      float v0 = s0[r] * scale + ((k0i <= q) ? 0.0f : -1e30f);
      float v1 = -1e30f;
      if (do2) {
        int k1i = kb + 16 + col;
        v1 = s1[r] * scale + ((k1i <= q) ? 0.0f : -1e30f);
      }
      float mx = fmaxf(v0, v1);
      mx = fmaxf(mx, __shfl_xor(mx, 1, 32));
      mx = fmaxf(mx, __shfl_xor(mx, 2, 32));
      mx = fmaxf(mx, __shfl_xor(mx, 4, 32));
      mx = fmaxf(mx, __shfl_xor(mx, 8, 32));
      float mnew = fmaxf(mrun[r], mx);
      float corr = __expf(mrun[r] - mnew);
      float p0 = __expf(v0 - mnew);
      float p1 = __expf(v1 - mnew);
      float rs = p0 + p1;
      rs += __shfl_xor(rs, 1, 32);
      rs += __shfl_xor(rs, 2, 32);
      rs += __shfl_xor(rs, 4, 32);
      rs += __shfl_xor(rs, 8, 32);
      lrun[r] = lrun[r] * corr + rs;
      mrun[r] = mnew;
#pragma unroll
      for (int f = 0; f < 8; f++) acc[f][r] *= corr;
      pbuf[(r + 8 * hlf) * 32 + col]      = (bf16_t)p0;
      pbuf[(r + 8 * hlf) * 32 + col + 16] = (bf16_t)p1;
    }

    v16bf pa = load_frag(pbuf, 0, 0, 32);
#pragma unroll
    for (int f = 0; f < 8; f++) {
      v16bf vb = load_frag(Vh, f * 16, kb, VT_LD);
      acc[f] = wmma_bf16(pa, vb, acc[f]);
    }
  }

#pragma unroll
  for (int f = 0; f < 8; f++) {
#pragma unroll
    for (int r = 0; r < 8; r++) {
      float o = acc[f][r] / lrun[r];
      int row = m0 + r + 8 * hlf;
      O[(size_t)row * HSZ + h * HD + f * 16 + col] = (bf16_t)o;
    }
  }
}

// ---------------------------------------------------------------------------
// Host launcher
// ---------------------------------------------------------------------------
extern "C" void kernel_launch(void* const* d_in, const int* in_sizes, int n_in,
                              void* d_out, int out_size, void* d_ws, size_t ws_size,
                              hipStream_t stream) {
  const float* hidden = (const float*)d_in[0];
  const int*   pos = (const int*)d_in[2];
  const float* w_q = (const float*)d_in[4];
  const float* w_k = (const float*)d_in[5];
  const float* w_v = (const float*)d_in[6];
  const float* w_o = (const float*)d_in[7];
  const float* qw  = (const float*)d_in[8];
  const float* kw  = (const float*)d_in[9];

  char* base = (char*)d_ws;
  size_t off = 0;
  auto alloc = [&](size_t bytes) -> void* {
    void* p = base + off;
    off += (bytes + 255) & ~(size_t)255;
    return p;
  };
  bf16_t* X    = (bf16_t*)alloc((size_t)S_LEN * HSZ * 2);
  bf16_t* Wqkv = (bf16_t*)alloc((size_t)NQKV * HSZ * 2);
  bf16_t* Wo   = (bf16_t*)alloc((size_t)HSZ * HSZ * 2);
  bf16_t* QKV  = (bf16_t*)alloc((size_t)S_LEN * NQKV * 2);
  bf16_t* Qnb  = (bf16_t*)alloc((size_t)S_LEN * HSZ * 2);
  bf16_t* Knb  = (bf16_t*)alloc((size_t)S_LEN * NKV * HD * 2);
  bf16_t* Vt   = (bf16_t*)alloc((size_t)NKV * HD * VT_LD * 2);
  bf16_t* AO   = (bf16_t*)alloc((size_t)S_LEN * HSZ * 2);
  (void)ws_size; (void)in_sizes; (void)n_in; (void)out_size;

  auto cvt = [&](const float* s, bf16_t* d, int n) {
    cvt_f32_bf16<<<(n + 255) / 256, 256, 0, stream>>>(s, d, n);
  };
  cvt(hidden, X, S_LEN * HSZ);
  cvt(w_q, Wqkv, NH * HD * HSZ);
  cvt(w_k, Wqkv + (size_t)NH * HD * HSZ, NKV * HD * HSZ);
  cvt(w_v, Wqkv + (size_t)(NH + NKV) * HD * HSZ, NKV * HD * HSZ);
  cvt(w_o, Wo, HSZ * HSZ);

  // QKV = X @ Wqkv^T   (bf16 out), TDM-fed 128x128 block tiles
  gemm_nt<false><<<dim3(NQKV / 128, S_LEN / 128), 256, 0, stream>>>(
      X, Wqkv, nullptr, QKV, S_LEN, NQKV, HSZ);

  norm_rope<<<(S_LEN * (NH + NKV)) / 8, 256, 0, stream>>>(QKV, qw, kw, pos,
                                                          Qnb, Knb);
  v_transpose<<<dim3((VT_LD + 255) / 256, HD, NKV), 256, 0, stream>>>(QKV, Vt);

  attn_fwd<<<dim3(S_LEN / 16, NH), 32, 0, stream>>>(Qnb, Knb, Vt, AO);

  // out = AO @ Wo^T  (fp32 out)
  gemm_nt<true><<<dim3(HSZ / 128, S_LEN / 128), 256, 0, stream>>>(
      AO, Wo, (float*)d_out, nullptr, S_LEN, HSZ, HSZ);
}
